// Attention_1563368095840
// MI455X (gfx1250) — matrix operand order repaired and checked by reading, not compiled, not run on hardware
//
#include <hip/hip_runtime.h>

typedef unsigned short u16;
typedef unsigned int   u32;
typedef unsigned long long u64;

typedef __attribute__((ext_vector_type(16))) __bf16 v16bf;
typedef __attribute__((ext_vector_type(8)))  float  v8f;
typedef __attribute__((ext_vector_type(8)))  u32    v8u;
typedef __attribute__((ext_vector_type(4)))  int    vint4;

// ---------- CDNA5 async global->LDS (direct DMA path, ASYNCcnt-tracked) ----------
#if __has_builtin(__builtin_amdgcn_global_load_async_to_lds_b128)
#define HAS_ASYNC_LDS 1
__device__ inline void async_g2l_b128(const void* g, void* l) {
    // Param types per hipcc diagnostic: (AS1 int4*, AS3 int4*, imm, imm).
    // generic->AS1 is value-preserving; generic LDS addr low 32 bits are the LDS
    // offset (ISA 10.2), so the int round-trip is exact.
    __builtin_amdgcn_global_load_async_to_lds_b128(
        (__attribute__((address_space(1))) vint4*)(u64)g,
        (__attribute__((address_space(3))) vint4*)(u32)(u64)l,
        0, 0);
}
__device__ inline void wait_async0() {
    asm volatile("s_wait_asynccnt 0" ::: "memory");
}
#else
#define HAS_ASYNC_LDS 0
#endif

// ---------- helpers ----------

__device__ inline u16 f2bf(float f) {  // round-to-nearest-even f32 -> bf16
    u32 u = __float_as_uint(f);
    u += 0x7fffu + ((u >> 16) & 1u);
    return (u16)(u >> 16);
}

// A-fragment (16x32, 16-bit): halves 0..7 at p, halves 8..15 at p+8 (u16 units).
__device__ inline v16bf ld_frag_a(const u16* p) {
    const u32* q0 = (const u32*)p;
    const u32* q1 = (const u32*)(p + 8);
    union { v8u u; v16bf f; } c;
    c.u[0] = q0[0]; c.u[1] = q0[1]; c.u[2] = q0[2]; c.u[3] = q0[3];
    c.u[4] = q1[0]; c.u[5] = q1[1]; c.u[6] = q1[2]; c.u[7] = q1[3];
    return c.f;
}

// B-fragment (32x16, 16-bit): 16 contiguous halves at p (buf stored [n][k]).
__device__ inline v16bf ld_frag_b(const u16* p) {
    const u32* q = (const u32*)p;
    union { v8u u; v16bf f; } c;
#pragma unroll
    for (int i = 0; i < 8; ++i) c.u[i] = q[i];
    return c.f;
}

__device__ inline v8f zero8() {
    v8f z;
#pragma unroll
    for (int i = 0; i < 8; ++i) z[i] = 0.0f;
    return z;
}

// ---------- conversion kernels ----------

__global__ void cast_bf16_k(const float* __restrict__ src, u16* __restrict__ dst, int n) {
    int i = blockIdx.x * 256 + threadIdx.x;
    if (i < n) dst[i] = f2bf(src[i]);
}

// W [16][1024][64] f32  ->  dst [k=1024][n=h*64+e] bf16
__global__ void pack_w_k(const float* __restrict__ W, u16* __restrict__ dst) {
    int i = blockIdx.x * 256 + threadIdx.x;   // over 1M
    int k = i >> 10, n = i & 1023;
    dst[i] = f2bf(W[((n >> 6) << 16) + (k << 6) + (n & 63)]);
}

// ---------- GEMM: C[M=4096][N=1024] = A[4096][1024] * B[1024][1024] + bias ----------

#define GP 72   // LDS k-pitch (64 + 8 pad)

template <int OUT_BF16>
__global__ __launch_bounds__(256) void gemm_k(const u16* __restrict__ A,
                                              const u16* __restrict__ B,
                                              const float* __restrict__ bias,
                                              u16* __restrict__ Cb,
                                              float* __restrict__ Cf) {
    __shared__ u16 As[128 * GP];   // [m][k]
    __shared__ u16 Bt[128 * GP];   // [n][k]  (transposed stage)

    const int tid  = threadIdx.x;
    const int lane = tid & 31;
    const int wv   = tid >> 5;       // 0..7
    const int mw   = (wv & 1) * 64;  // wave M base in tile
    const int nw   = (wv >> 1) * 32; // wave N base in tile
    const int l15  = lane & 15;
    const int hi   = lane >> 4;
    const int hi8  = hi << 3;
    const int hi16 = hi << 4;

    const int rowB = blockIdx.y * 128;
    const int colB = blockIdx.x * 128;

    v8f acc[4][2];
#pragma unroll
    for (int mi = 0; mi < 4; ++mi)
#pragma unroll
        for (int ni = 0; ni < 2; ++ni) acc[mi][ni] = zero8();

    const u32* Ag = (const u32*)A;
    const u32* Bg = (const u32*)B;

    // register double-buffer for the transposed B stage
    u32 rb[16];
#pragma unroll
    for (int i = 0; i < 16; ++i) {
        int idx = tid + i * 256;
        rb[i] = Bg[((idx >> 6)) * 512 + (colB >> 1) + (idx & 63)];
    }

    for (int k0 = 0; k0 < 1024; k0 += 64) {
        // ---- stage A tile 128x64 halves ----
#if HAS_ASYNC_LDS
#pragma unroll
        for (int i = 0; i < 4; ++i) {
            int idx = tid + i * 256;          // 1024 x b128 chunks
            int r = idx >> 3, c = idx & 7;
            async_g2l_b128(&A[(rowB + r) * 1024 + k0 + c * 8], &As[r * GP + c * 8]);
        }
#else
#pragma unroll
        for (int i = 0; i < 16; ++i) {
            int idx = tid + i * 256;
            int r = idx >> 5, kd = idx & 31;
            u32 d = Ag[(rowB + r) * 512 + (k0 >> 1) + kd];
            *(u32*)&As[r * GP + kd * 2] = d;
        }
#endif
        // ---- stage B tile (transpose from registers) ----
#pragma unroll
        for (int i = 0; i < 16; ++i) {
            int idx = tid + i * 256;
            int kr = idx >> 6, nd = idx & 63;
            int n = nd * 2;
            Bt[n * GP + kr]       = (u16)rb[i];
            Bt[(n + 1) * GP + kr] = (u16)(rb[i] >> 16);
        }
        // prefetch next B tile into registers while this tile computes
        if (k0 + 64 < 1024) {
#pragma unroll
            for (int i = 0; i < 16; ++i) {
                int idx = tid + i * 256;
                rb[i] = Bg[(k0 + 64 + (idx >> 6)) * 512 + (colB >> 1) + (idx & 63)];
            }
        }
#if HAS_ASYNC_LDS
        wait_async0();
#endif
        __syncthreads();

#pragma unroll
        for (int kk = 0; kk < 64; kk += 32) {
            v16bf af[4], bf[2];
#pragma unroll
            for (int mi = 0; mi < 4; ++mi)
                af[mi] = ld_frag_a(&As[(mw + mi * 16 + l15) * GP + kk + hi16]);
#pragma unroll
            for (int ni = 0; ni < 2; ++ni)
                bf[ni] = ld_frag_b(&Bt[(nw + ni * 16 + l15) * GP + kk + hi16]);
#pragma unroll
            for (int mi = 0; mi < 4; ++mi)
#pragma unroll
                for (int ni = 0; ni < 2; ++ni)
                    acc[mi][ni] = __builtin_amdgcn_wmma_f32_16x16x32_bf16(
                        false, af[mi], false, bf[ni], (short)0, acc[mi][ni], false, false);
        }
        __syncthreads();
    }

#pragma unroll
    for (int mi = 0; mi < 4; ++mi) {
#pragma unroll
        for (int ni = 0; ni < 2; ++ni) {
            int row0 = rowB + mw + mi * 16 + hi8;
            int col  = colB + nw + ni * 16 + l15;
            float bv = bias[col];
#pragma unroll
            for (int r = 0; r < 8; ++r) {
                float v = acc[mi][ni][r] + bv;
                if (OUT_BF16) Cb[(row0 + r) * 1024 + col] = f2bf(v);
                else          Cf[(row0 + r) * 1024 + col] = v;
            }
        }
    }
}

// ---------- attention: flash-style, per (b,h), 64 q rows per block ----------

#define KP 72   // Kld pitch (64 dims + 8)
#define VP 40   // Vt / P pitch (32 + 8)

__global__ __launch_bounds__(128) void attn_k(const u16* __restrict__ Q,
                                              const u16* __restrict__ K,
                                              const u16* __restrict__ V,
                                              u16* __restrict__ Z) {
    __shared__ u16 Kld[32 * KP];     // [key][dim]
    __shared__ u16 Vt[64 * VP];      // [dim][key]
    __shared__ u16 Pld[4][16 * VP];  // per-wave P staging [row][key]

    const int tid  = threadIdx.x;
    const int lane = tid & 31;
    const int w    = tid >> 5;       // 0..3
    const int l15  = lane & 15;
    const int hi   = lane >> 4;
    const int hi8  = hi << 3;
    const int hi16 = hi << 4;

    const int b  = blockIdx.z;
    const int h  = blockIdx.y;
    const int qt = blockIdx.x * 64 + w * 16;

    const u16* qbase = Q + (size_t)(b * 2048 + qt + l15) * 1024 + h * 64;
    v16bf qa0 = ld_frag_a(qbase + 0  + hi16);
    v16bf qa1 = ld_frag_a(qbase + 32 + hi16);

    float mrow[8], lrow[8];
    v8f o[4];
#pragma unroll
    for (int r = 0; r < 8; ++r) { mrow[r] = -1e30f; lrow[r] = 0.0f; }
#pragma unroll
    for (int t = 0; t < 4; ++t) o[t] = zero8();

    const u32* Vg = (const u32*)V;
#if !HAS_ASYNC_LDS
    const u32* Kg = (const u32*)K;
#endif
    const int kbMax = blockIdx.x * 2 + 2;   // 32-key blocks covering causal range

    for (int kb = 0; kb < kbMax; ++kb) {
        const int kBase = kb * 32;
        // ---- stage K block [32 keys][64 dims] ----
#if HAS_ASYNC_LDS
#pragma unroll
        for (int i = 0; i < 2; ++i) {
            int idx = tid + i * 128;          // 256 x b128 chunks
            int key = idx >> 3, c = idx & 7;
            async_g2l_b128(&K[(size_t)(b * 2048 + kBase + key) * 1024 + h * 64 + c * 8],
                           &Kld[key * KP + c * 8]);
        }
#else
#pragma unroll
        for (int i = 0; i < 8; ++i) {
            int idx = tid + i * 128;
            int key = idx >> 5, kd = idx & 31;
            u32 d = Kg[(b * 2048 + kBase + key) * 512 + h * 32 + kd];
            *(u32*)&Kld[key * KP + kd * 2] = d;
        }
#endif
        // ---- stage V block transposed -> Vt[dim][key] ----
#pragma unroll
        for (int i = 0; i < 8; ++i) {
            int idx = tid + i * 128;
            int key = idx >> 5, dd = idx & 31;
            u32 d = Vg[(b * 2048 + kBase + key) * 512 + h * 32 + dd];
            Vt[(2 * dd) * VP + key]     = (u16)d;
            Vt[(2 * dd + 1) * VP + key] = (u16)(d >> 16);
        }
        // prefetch next block's K/V rows into cache (global_prefetch_b8)
        if (kb + 1 < kbMax) {
            int row = b * 2048 + kBase + 32 + (tid & 31);
            __builtin_prefetch(&K[(size_t)row * 1024 + h * 64], 0, 1);
            __builtin_prefetch(&V[(size_t)row * 1024 + h * 64], 0, 1);
        }
#if HAS_ASYNC_LDS
        wait_async0();
#endif
        __syncthreads();

        if (kBase <= qt + 15) {
            const bool act1 = (kBase + 16 <= qt + 15);

            // ---- S tiles: 16 q x 32 keys ----
            v8f s0 = zero8(), s1 = zero8();
            {
                v16bf b0 = ld_frag_b(&Kld[(0  + l15) * KP + 0  + hi16]);
                v16bf b1 = ld_frag_b(&Kld[(0  + l15) * KP + 32 + hi16]);
                s0 = __builtin_amdgcn_wmma_f32_16x16x32_bf16(false, qa0, false, b0, (short)0, s0, false, false);
                s0 = __builtin_amdgcn_wmma_f32_16x16x32_bf16(false, qa1, false, b1, (short)0, s0, false, false);
            }
            if (act1) {
                v16bf b0 = ld_frag_b(&Kld[(16 + l15) * KP + 0  + hi16]);
                v16bf b1 = ld_frag_b(&Kld[(16 + l15) * KP + 32 + hi16]);
                s1 = __builtin_amdgcn_wmma_f32_16x16x32_bf16(false, qa0, false, b0, (short)0, s1, false, false);
                s1 = __builtin_amdgcn_wmma_f32_16x16x32_bf16(false, qa1, false, b1, (short)0, s1, false, false);
            }

            // scale + causal mask
#pragma unroll
            for (int r = 0; r < 8; ++r) {
                int qrow = qt + r + hi8;
                s0[r] = (kBase + l15      <= qrow) ? s0[r] * 0.125f : -1e30f;
                s1[r] = (kBase + 16 + l15 <= qrow) ? s1[r] * 0.125f : -1e30f;
            }

            // ---- online softmax (row lives in a 16-lane half) ----
            float mnew[8], corr[8];
#pragma unroll
            for (int r = 0; r < 8; ++r) {
                float v = fmaxf(s0[r], s1[r]);
#pragma unroll
                for (int m = 1; m < 16; m <<= 1) v = fmaxf(v, __shfl_xor(v, m, 32));
                mnew[r] = fmaxf(mrow[r], v);
                corr[r] = __expf(mrow[r] - mnew[r]);
                mrow[r] = mnew[r];
                s0[r] = __expf(s0[r] - mnew[r]);
                s1[r] = act1 ? __expf(s1[r] - mnew[r]) : 0.0f;
                float sum = s0[r] + s1[r];
#pragma unroll
                for (int m = 1; m < 16; m <<= 1) sum += __shfl_xor(sum, m, 32);
                lrow[r] = lrow[r] * corr[r] + sum;
            }
#pragma unroll
            for (int t = 0; t < 4; ++t)
#pragma unroll
                for (int r = 0; r < 8; ++r) o[t][r] *= corr[r];

            // ---- P: C-layout -> A-layout via wave-private LDS ----
#pragma unroll
            for (int r = 0; r < 8; ++r) {
                int row = r + hi8;
                Pld[w][row * VP + l15]      = f2bf(s0[r]);
                Pld[w][row * VP + 16 + l15] = f2bf(s1[r]);
            }
            v16bf pa = ld_frag_a(&Pld[w][l15 * VP + hi16]);

            // ---- O += P * V ----
#pragma unroll
            for (int ni = 0; ni < 4; ++ni) {
                v16bf vb = ld_frag_b(&Vt[(ni * 16 + l15) * VP + hi16]);
                o[ni] = __builtin_amdgcn_wmma_f32_16x16x32_bf16(
                    false, pa, false, vb, (short)0, o[ni], false, false);
            }
        }
        __syncthreads();
    }

    // normalize + store Z (bf16)
#pragma unroll
    for (int ni = 0; ni < 4; ++ni) {
#pragma unroll
        for (int r = 0; r < 8; ++r) {
            int row = qt + r + hi8;
            float v = o[ni][r] / lrow[r];
            Z[(size_t)(b * 2048 + row) * 1024 + h * 64 + ni * 16 + l15] = f2bf(v);
        }
    }
}

// ---------- host launch ----------

extern "C" void kernel_launch(void* const* d_in, const int* in_sizes, int n_in,
                              void* d_out, int out_size, void* d_ws, size_t ws_size,
                              hipStream_t stream) {
    const float* x  = (const float*)d_in[0];
    const float* WQ = (const float*)d_in[1];
    const float* WK = (const float*)d_in[2];
    const float* WV = (const float*)d_in[3];
    const float* WO = (const float*)d_in[4];
    const float* bQ = (const float*)d_in[5];
    const float* bK = (const float*)d_in[6];
    const float* bV = (const float*)d_in[7];
    const float* bO = (const float*)d_in[8];

    // workspace layout (u16 elements): 48 MB total
    u16* xb = (u16*)d_ws;           // 4M elems (8 MB)
    u16* wq = xb + (4 << 20);       // 1M each
    u16* wk = wq + (1 << 20);
    u16* wv = wk + (1 << 20);
    u16* wo = wv + (1 << 20);
    u16* Qb = wo + (1 << 20);       // 4M each
    u16* Kb = Qb + (4 << 20);
    u16* Vb = Kb + (4 << 20);
    u16* Zb = Vb + (4 << 20);

    cast_bf16_k<<<16384, 256, 0, stream>>>(x, xb, 4194304);
    pack_w_k<<<4096, 256, 0, stream>>>(WQ, wq);
    pack_w_k<<<4096, 256, 0, stream>>>(WK, wk);
    pack_w_k<<<4096, 256, 0, stream>>>(WV, wv);
    cast_bf16_k<<<4096, 256, 0, stream>>>(WO, wo, 1048576);

    dim3 gg(8, 32);   // N/128, M/128
    gemm_k<1><<<gg, 256, 0, stream>>>(xb, wq, bQ, Qb, nullptr);
    gemm_k<1><<<gg, 256, 0, stream>>>(xb, wk, bK, Kb, nullptr);
    gemm_k<1><<<gg, 256, 0, stream>>>(xb, wv, bV, Vb, nullptr);

    attn_k<<<dim3(32, 16, 2), 128, 0, stream>>>(Qb, Kb, Vb, Zb);

    gemm_k<0><<<gg, 256, 0, stream>>>(Zb, wo, bO, nullptr, (float*)d_out);
}
